// _BidirectionalMamba2Block_88897233092707
// MI455X (gfx1250) — compile-verified, matrix operand order checked by reading, not run on hardware
//
#include <hip/hip_runtime.h>
#include <hip/hip_bf16.h>

// ---------------------------------------------------------------------------
// Bidirectional Mamba2 block for MI455X (gfx1250, wave32, WMMA).
// GEMMs: v_wmma_f32_16x16x32_bf16 + async global->LDS double-buffered B panel.
// Scan: register-resident VALU with LDS staging + wave32 shuffles.
// ---------------------------------------------------------------------------

#define D_MODEL   768
#define D_STATE   64
#define D_INNER   1536
#define HEADDIM   64
#define NHEADS    24
#define D_CONV    4
#define CONV_DIM  (D_INNER + 2 * D_STATE)              // 1664
#define D_IN_PROJ (2 * D_INNER + 2 * D_STATE + NHEADS) // 3224
#define BB        4
#define LL        2048

typedef __attribute__((ext_vector_type(16))) __bf16 v16bf;
typedef __attribute__((ext_vector_type(8)))  __bf16 v8bf;
typedef __attribute__((ext_vector_type(8)))  float  v8f;

typedef int v4i __attribute__((vector_size(16)));
typedef __attribute__((address_space(1))) v4i* as1_v4i;   // global (__device__)
typedef __attribute__((address_space(3))) v4i* as3_v4i;   // LDS (__shared__)

__device__ __forceinline__ float wave_sum(float v) {
    v += __shfl_xor(v, 1, 32);
    v += __shfl_xor(v, 2, 32);
    v += __shfl_xor(v, 4, 32);
    v += __shfl_xor(v, 8, 32);
    v += __shfl_xor(v, 16, 32);
    return v;
}

__device__ __forceinline__ float silu_f(float v) {
    return v / (1.0f + __expf(-v));
}

// gfx1250 async global->LDS copy of 16 bytes (ASYNCcnt-tracked).
__device__ __forceinline__ void async_copy_b128(const __bf16* g, __bf16* l) {
#if __has_builtin(__builtin_amdgcn_global_load_async_to_lds_b128)
    __builtin_amdgcn_global_load_async_to_lds_b128(
        (as1_v4i)(v4i*)g, (as3_v4i)(v4i*)l, 0, 0);
#else
    asm volatile("global_load_async_to_lds_b128 %0, %1, off"
                 :: "v"((unsigned)(unsigned long long)l),
                    "v"((unsigned long long)g)
                 : "memory");
#endif
}

__device__ __forceinline__ void wait_async0() {
#if __has_builtin(__builtin_amdgcn_s_wait_asynccnt)
    __builtin_amdgcn_s_wait_asynccnt(0);
#else
    asm volatile("s_wait_asynccnt 0x0" ::: "memory");
#endif
}

__device__ __forceinline__ v16bf load_frag(const __bf16* p) {
    v8bf lo = *(const v8bf*)p;
    v8bf hi = *(const v8bf*)(p + 16);
    v16bf r;
#pragma unroll
    for (int i = 0; i < 8; ++i) { r[i] = lo[i]; r[8 + i] = hi[i]; }
    return r;
}

// --------------------------- elementwise helpers ---------------------------

__global__ void k_f32_to_bf16(const float* __restrict__ in, __bf16* __restrict__ out, size_t n) {
    size_t i = (size_t)blockIdx.x * blockDim.x + threadIdx.x;
    if (i < n) out[i] = (__bf16)in[i];
}

// x [B,L,D_MODEL] f32 -> bf16, optionally length-flipped along L.
__global__ void k_flip_convert(const float* __restrict__ x, const int* __restrict__ lengths,
                               __bf16* __restrict__ out, int flip) {
    size_t idx = (size_t)blockIdx.x * blockDim.x + threadIdx.x;
    if (idx >= (size_t)BB * LL * D_MODEL) return;
    int d = (int)(idx % D_MODEL);
    size_t bl = idx / D_MODEL;
    int l = (int)(bl % LL);
    int b = (int)(bl / LL);
    int src = l;
    if (flip) {
        int Ln = lengths[b];
        src = (l < Ln) ? (Ln - 1 - l) : l;
    }
    out[idx] = (__bf16)x[((size_t)b * LL + src) * D_MODEL + d];
}

// depthwise causal conv width 4 over xBC slice of zxbcdt, then SiLU.
__global__ void k_conv_silu(const float* __restrict__ zx, const float* __restrict__ conv_w,
                            const float* __restrict__ conv_b, float* __restrict__ xbc) {
    size_t idx = (size_t)blockIdx.x * blockDim.x + threadIdx.x;
    if (idx >= (size_t)BB * LL * CONV_DIM) return;
    int c = (int)(idx % CONV_DIM);
    size_t bl = idx / CONV_DIM;
    int l = (int)(bl % LL);
    float acc = conv_b[c];
#pragma unroll
    for (int k = 0; k < D_CONV; ++k) {
        int shift = (D_CONV - 1) - k;              // xp[l - shift]
        if (l >= shift)
            acc += zx[(bl - shift) * D_IN_PROJ + D_INNER + c] * conv_w[c * D_CONV + k];
    }
    xbc[idx] = silu_f(acc);
}

__global__ void k_dt_softplus(const float* __restrict__ zx, const float* __restrict__ dt_bias,
                              float* __restrict__ dt) {
    size_t idx = (size_t)blockIdx.x * blockDim.x + threadIdx.x;
    if (idx >= (size_t)BB * LL * NHEADS) return;
    int h = (int)(idx % NHEADS);
    size_t row = idx / NHEADS;
    float v = zx[row * D_IN_PROJ + (D_INNER + CONV_DIM) + h] + dt_bias[h];
    dt[idx] = (v > 20.0f) ? v : log1pf(__expf(v));
}

// ------------------------------ WMMA GEMM ----------------------------------
// C[M,N] f32 = A[M,K] bf16 (row major) x Bw[N,K] bf16 (row major, i.e. B^T).
// Block tile: 128(M) x 64(N); 8 waves, one 16x64 strip per wave.
// B panel staged in LDS via async global->LDS copies, double-buffered in
// 64-wide K chunks; shared by all 8 waves (8x global-traffic reduction).
__global__ __launch_bounds__(256) void k_gemm_bf16_wmma(
    const __bf16* __restrict__ A, const __bf16* __restrict__ Bw,
    float* __restrict__ C, int M, int N, int K) {
    const int tid   = threadIdx.x;
    const int lane  = tid & 31;
    const int wave  = tid >> 5;
    const int mtile = blockIdx.y * 128 + wave * 16;
    const int nbase = blockIdx.x * 64;

    // A fragment addressing: lane holds row m = mtile + (lane&15),
    // K-halves {kb..kb+7} and {16+kb..16+kb+7}, kb = (lane>=16)*8.
    int mrow = mtile + (lane & 15);
    if (mrow >= M) mrow = M - 1;                   // clamp (stores guarded)
    const int kb = (lane >> 4) * 8;
    const __bf16* arow = A + (size_t)mrow * K + kb;

    // LDS B panel: [buf][n][k], row stride 72 elems (144B, 16B-aligned).
    __shared__ __bf16 Bs[2][64][72];

    // staging map: 512 16B-chunks per 64x64 tile; thread covers chunks
    // tid and tid+256 -> n = (tid>>3)+{0,32}, sub = tid&7 (8 chunks/row).
    const int sn  = tid >> 3;
    const int ssb = tid & 7;
    int ng0 = nbase + sn;       if (ng0 >= N) ng0 = N - 1;
    int ng1 = nbase + sn + 32;  if (ng1 >= N) ng1 = N - 1;
    const __bf16* brow0 = Bw + (size_t)ng0 * K + ssb * 8;
    const __bf16* brow1 = Bw + (size_t)ng1 * K + ssb * 8;

    v8f acc[4];
#pragma unroll
    for (int j = 0; j < 4; ++j)
#pragma unroll
        for (int i = 0; i < 8; ++i) acc[j][i] = 0.0f;

    const int NC = K >> 6;                          // 64-wide K chunks

    // prime chunk 0 into buffer 0
    async_copy_b128(brow0, &Bs[0][sn][ssb * 8]);
    async_copy_b128(brow1, &Bs[0][sn + 32][ssb * 8]);

    for (int c = 0; c < NC; ++c) {
        wait_async0();
        __syncthreads();                            // chunk c resident in buf c&1

        if (c + 1 < NC) {                           // stream chunk c+1 -> other buf
            const int kn = (c + 1) * 64;
            async_copy_b128(brow0 + kn, &Bs[(c + 1) & 1][sn][ssb * 8]);
            async_copy_b128(brow1 + kn, &Bs[(c + 1) & 1][sn + 32][ssb * 8]);
        }

        const int kc = c * 64;
        __builtin_prefetch(arow + kc + 128, 0, 1);  // global_prefetch_b8 (A stream)

#pragma unroll
        for (int s = 0; s < 2; ++s) {
            const int ks = s * 32;
            v16bf a = load_frag(arow + kc + ks);
            // gather all 4 B fragments first, then back-to-back WMMAs
            v16bf b0 = load_frag(&Bs[c & 1][ 0 + (lane & 15)][ks + kb]);
            v16bf b1 = load_frag(&Bs[c & 1][16 + (lane & 15)][ks + kb]);
            v16bf b2 = load_frag(&Bs[c & 1][32 + (lane & 15)][ks + kb]);
            v16bf b3 = load_frag(&Bs[c & 1][48 + (lane & 15)][ks + kb]);
            acc[0] = __builtin_amdgcn_wmma_f32_16x16x32_bf16(false, a, false, b0, (short)0, acc[0], false, false);
            acc[1] = __builtin_amdgcn_wmma_f32_16x16x32_bf16(false, a, false, b1, (short)0, acc[1], false, false);
            acc[2] = __builtin_amdgcn_wmma_f32_16x16x32_bf16(false, a, false, b2, (short)0, acc[2], false, false);
            acc[3] = __builtin_amdgcn_wmma_f32_16x16x32_bf16(false, a, false, b3, (short)0, acc[3], false, false);
        }
    }

    // D layout: lanes 0-15 -> M=r, lanes 16-31 -> M=8+r; N = lane&15.
    const int crow0 = mtile + ((lane >> 4) ? 8 : 0);
#pragma unroll
    for (int j = 0; j < 4; ++j) {
        int col = nbase + j * 16 + (lane & 15);
        if (col < N && crow0 + 7 < M) {
#pragma unroll
            for (int r = 0; r < 8; ++r)
                C[(size_t)(crow0 + r) * N + col] = acc[j][r];
        }
    }
}

// ------------------------------ selective scan -----------------------------
// One block per (batch, head). 256 threads: thread t owns p=t>>2, n in
// [(t&3)*16, +16). State h[64,64] lives in registers (16 floats / thread).
__global__ __launch_bounds__(256) void k_scan(
    const float* __restrict__ xbc, const float* __restrict__ dt,
    const float* __restrict__ A_log, const float* __restrict__ Dp,
    float* __restrict__ y) {
    const int h = blockIdx.x % NHEADS;
    const int b = blockIdx.x / NHEADS;
    const int t  = threadIdx.x;
    const int p  = t >> 2;
    const int nq = t & 3;
    const int n0 = nq * 16;

    const float A  = -__expf(A_log[h]);
    const float Dh = Dp[h];

    float hs[16];
#pragma unroll
    for (int i = 0; i < 16; ++i) hs[i] = 0.0f;

    __shared__ float Bs[D_STATE];
    __shared__ float Cs[D_STATE];

    for (int l = 0; l < LL; ++l) {
        const size_t row = (size_t)b * LL + l;
        if (t < 64)        Bs[t]      = xbc[row * CONV_DIM + D_INNER + t];
        else if (t < 128)  Cs[t - 64] = xbc[row * CONV_DIM + D_INNER + D_STATE + (t - 64)];
        __syncthreads();

        const float dtt = dt[row * NHEADS + h];
        const float dA  = __expf(dtt * A);
        const float xp  = xbc[row * CONV_DIM + h * HEADDIM + p];
        const float com = dtt * xp;

        float ysum = 0.0f;
#pragma unroll
        for (int i = 0; i < 16; ++i) {
            hs[i] = hs[i] * dA + com * Bs[n0 + i];
            ysum += hs[i] * Cs[n0 + i];
        }
        ysum += __shfl_xor(ysum, 1, 32);
        ysum += __shfl_xor(ysum, 2, 32);
        if (nq == 0) y[row * D_INNER + h * HEADDIM + p] = ysum + Dh * xp;
        __syncthreads();
    }
}

// gated RMSNorm -> bf16; one block per row.
__global__ __launch_bounds__(256) void k_rms_gate(
    const float* __restrict__ y, const float* __restrict__ zx,
    const float* __restrict__ norm_w, __bf16* __restrict__ out) {
    const size_t row = blockIdx.x;
    const int t = threadIdx.x;
    float v[6];
    float ss = 0.0f;
#pragma unroll
    for (int j = 0; j < 6; ++j) {
        int e = t + j * 256;
        float z = zx[row * D_IN_PROJ + e];
        float g = y[row * D_INNER + e] * silu_f(z);
        v[j] = g;
        ss += g * g;
    }
    __shared__ float red[8];
    float ws = wave_sum(ss);
    if ((t & 31) == 0) red[t >> 5] = ws;
    __syncthreads();
    float tot = 0.0f;
#pragma unroll
    for (int w = 0; w < 8; ++w) tot += red[w];
    const float scale = rsqrtf(tot / (float)D_INNER + 1e-5f);
#pragma unroll
    for (int j = 0; j < 6; ++j) {
        int e = t + j * 256;
        out[row * D_INNER + e] = (__bf16)(v[j] * scale * norm_w[e]);
    }
}

// comb = concat(fwd, flip(bwd_f)) -> bf16 [B*L, 2*D_MODEL]
__global__ void k_comb(const float* __restrict__ fwd, const float* __restrict__ bwdf,
                       const int* __restrict__ lengths, __bf16* __restrict__ comb) {
    size_t idx = (size_t)blockIdx.x * blockDim.x + threadIdx.x;
    if (idx >= (size_t)BB * LL * (2 * D_MODEL)) return;
    int e = (int)(idx % (2 * D_MODEL));
    size_t row = idx / (2 * D_MODEL);
    int l = (int)(row % LL);
    int b = (int)(row / LL);
    float val;
    if (e < D_MODEL) {
        val = fwd[row * D_MODEL + e];
    } else {
        int Ln = lengths[b];
        int src = (l < Ln) ? (Ln - 1 - l) : l;
        val = bwdf[((size_t)b * LL + src) * D_MODEL + (e - D_MODEL)];
    }
    comb[idx] = (__bf16)val;
}

// out = LayerNorm(lin + op_b + x) ; one block per row.
__global__ __launch_bounds__(256) void k_final_ln(
    const float* __restrict__ lin, const float* __restrict__ x,
    const float* __restrict__ op_b, const float* __restrict__ ln_g,
    const float* __restrict__ ln_b, float* __restrict__ out) {
    const size_t row = blockIdx.x;
    const int t = threadIdx.x;
    float v[3];
    float s = 0.0f, s2 = 0.0f;
#pragma unroll
    for (int j = 0; j < 3; ++j) {
        int e = t + j * 256;
        float r = lin[row * D_MODEL + e] + op_b[e] + x[row * D_MODEL + e];
        v[j] = r;
        s += r;
        s2 += r * r;
    }
    __shared__ float redA[8], redB[8];
    float wsA = wave_sum(s), wsB = wave_sum(s2);
    if ((t & 31) == 0) { redA[t >> 5] = wsA; redB[t >> 5] = wsB; }
    __syncthreads();
    float ts = 0.0f, ts2 = 0.0f;
#pragma unroll
    for (int w = 0; w < 8; ++w) { ts += redA[w]; ts2 += redB[w]; }
    const float mu  = ts / (float)D_MODEL;
    const float var = ts2 / (float)D_MODEL - mu * mu;
    const float inv = rsqrtf(var + 1e-5f);
#pragma unroll
    for (int j = 0; j < 3; ++j) {
        int e = t + j * 256;
        out[row * D_MODEL + e] = (v[j] - mu) * inv * ln_g[e] + ln_b[e];
    }
}

// ------------------------------- launch ------------------------------------

static inline size_t cdiv(size_t a, size_t b) { return (a + b - 1) / b; }

extern "C" void kernel_launch(void* const* d_in, const int* in_sizes, int n_in,
                              void* d_out, int out_size, void* d_ws, size_t ws_size,
                              hipStream_t stream) {
    (void)in_sizes; (void)n_in; (void)out_size; (void)ws_size;
    const float* x        = (const float*)d_in[0];
    const int*   lengths  = (const int*)d_in[1];
    const float* in_w[2]   = { (const float*)d_in[2],  (const float*)d_in[10] };
    const float* conv_w[2] = { (const float*)d_in[3],  (const float*)d_in[11] };
    const float* conv_b[2] = { (const float*)d_in[4],  (const float*)d_in[12] };
    const float* dt_bias[2]= { (const float*)d_in[5],  (const float*)d_in[13] };
    const float* A_log[2]  = { (const float*)d_in[6],  (const float*)d_in[14] };
    const float* Dp[2]     = { (const float*)d_in[7],  (const float*)d_in[15] };
    const float* norm_w[2] = { (const float*)d_in[8],  (const float*)d_in[16] };
    const float* out_w[2]  = { (const float*)d_in[9],  (const float*)d_in[17] };
    const float* op_w  = (const float*)d_in[18];
    const float* op_b  = (const float*)d_in[19];
    const float* ln_g  = (const float*)d_in[20];
    const float* ln_b  = (const float*)d_in[21];

    const size_t M = (size_t)BB * LL;              // 8192 rows

    // workspace carve-up (256B aligned)
    char* ws = (char*)d_ws;
    size_t off = 0;
    auto carve = [&](size_t bytes) -> void* {
        void* p = ws + off;
        off += (bytes + 255) & ~(size_t)255;
        return p;
    };
    __bf16* inw_bf[2]  = { (__bf16*)carve((size_t)D_IN_PROJ * D_MODEL * 2),
                           (__bf16*)carve((size_t)D_IN_PROJ * D_MODEL * 2) };
    __bf16* outw_bf[2] = { (__bf16*)carve((size_t)D_MODEL * D_INNER * 2),
                           (__bf16*)carve((size_t)D_MODEL * D_INNER * 2) };
    __bf16* opw_bf     = (__bf16*)carve((size_t)D_MODEL * (2 * D_MODEL) * 2);
    __bf16* xb         = (__bf16*)carve(M * D_MODEL * 2);
    float*  zx         = (float*) carve(M * D_IN_PROJ * 4);
    float*  xbc        = (float*) carve(M * CONV_DIM * 4);
    float*  dtb        = (float*) carve(M * NHEADS * 4);
    float*  yb         = (float*) carve(M * D_INNER * 4);
    __bf16* ybf        = (__bf16*)carve(M * D_INNER * 2);
    float*  dirout[2]  = { (float*)carve(M * D_MODEL * 4),
                           (float*)carve(M * D_MODEL * 4) };
    __bf16* comb       = (__bf16*)carve(M * (2 * D_MODEL) * 2);
    float*  lin        = (float*) carve(M * D_MODEL * 4);

    const int T = 256;

    // weights -> bf16 (small)
    for (int d = 0; d < 2; ++d) {
        size_t n1 = (size_t)D_IN_PROJ * D_MODEL;
        k_f32_to_bf16<<<cdiv(n1, T), T, 0, stream>>>(in_w[d], inw_bf[d], n1);
        size_t n2 = (size_t)D_MODEL * D_INNER;
        k_f32_to_bf16<<<cdiv(n2, T), T, 0, stream>>>(out_w[d], outw_bf[d], n2);
    }
    {
        size_t n3 = (size_t)D_MODEL * 2 * D_MODEL;
        k_f32_to_bf16<<<cdiv(n3, T), T, 0, stream>>>(op_w, opw_bf, n3);
    }

    for (int d = 0; d < 2; ++d) {
        // input (bf16, flipped for backward dir)
        k_flip_convert<<<cdiv(M * D_MODEL, T), T, 0, stream>>>(x, lengths, xb, d);

        // in_proj: [8192,3224] = [8192,768] x [3224,768]^T
        dim3 g1((unsigned)cdiv(D_IN_PROJ, 64), (unsigned)cdiv(M, 128));
        k_gemm_bf16_wmma<<<g1, T, 0, stream>>>(xb, inw_bf[d], zx, (int)M, D_IN_PROJ, D_MODEL);

        k_conv_silu<<<cdiv(M * CONV_DIM, T), T, 0, stream>>>(zx, conv_w[d], conv_b[d], xbc);
        k_dt_softplus<<<cdiv(M * NHEADS, T), T, 0, stream>>>(zx, dt_bias[d], dtb);

        k_scan<<<BB * NHEADS, T, 0, stream>>>(xbc, dtb, A_log[d], Dp[d], yb);

        k_rms_gate<<<(unsigned)M, T, 0, stream>>>(yb, zx, norm_w[d], ybf);

        // out_proj: [8192,768] = [8192,1536] x [768,1536]^T
        dim3 g2((unsigned)cdiv(D_MODEL, 64), (unsigned)cdiv(M, 128));
        k_gemm_bf16_wmma<<<g2, T, 0, stream>>>(ybf, outw_bf[d], dirout[d], (int)M, D_MODEL, D_INNER);
    }

    // combine + final projection + layernorm
    k_comb<<<cdiv(M * 2 * D_MODEL, T), T, 0, stream>>>(dirout[0], dirout[1], lengths, comb);
    dim3 g3((unsigned)cdiv(D_MODEL, 64), (unsigned)cdiv(M, 128));
    k_gemm_bf16_wmma<<<g3, T, 0, stream>>>(comb, opw_bf, lin, (int)M, D_MODEL, 2 * D_MODEL);
    k_final_ln<<<(unsigned)M, T, 0, stream>>>(lin, x, op_b, ln_g, ln_b, (float*)d_out);
}
